// GCN2_38817914421894
// MI455X (gfx1250) — matrix operand order repaired and checked by reading, not compiled, not run on hardware
//
#include <hip/hip_runtime.h>
#include <hip/hip_bf16.h>

// ---------------------------------------------------------------------------
// GCNII (2-layer GCN2Conv) for MI455X / gfx1250, wave32.
//
//   1) weights -> bf16, transposed to [N][K] (TDM/K-contiguous fragment order)
//   2) deg histogram (u32 atomics) -> dinv = rsqrt(deg+1)
//   3) h0 = relu(x @ w1 + b1)       WMMA bf16; A: f32->bf16 fused staging,
//                                   W: TDM-fed double-buffered tiles
//   4) 2x: agg = scatter-add(norm*h[row])   wave/edge, global_atomic_add_f32
//          Abf = bf16(0.9*agg + 0.1*h0)     bandwidth pre-pass
//          h   = Abf @ conv_w[l]            TDM-fed double-buffered WMMA
//   5) logits = bf16(h) @ w2 + b2           TDM-fed WMMA
// ---------------------------------------------------------------------------

typedef __attribute__((ext_vector_type(16))) __bf16 v16bf;
typedef __attribute__((ext_vector_type(8)))  float  v8f;
typedef __attribute__((ext_vector_type(4)))  unsigned int u32x4;
typedef __attribute__((ext_vector_type(4)))  int i32x4;
typedef __attribute__((ext_vector_type(8)))  int i32x8;

union Frag16 { v16bf v; uint4 q[2]; };

__device__ __forceinline__ unsigned short f32_to_bf16(float f) {
    unsigned int u = __float_as_uint(f);
    unsigned int r = u + 0x7FFFu + ((u >> 16) & 1u);   // round-to-nearest-even
    return (unsigned short)(r >> 16);
}

__device__ __forceinline__ void atomAddF(float* p, float v) {
    __hip_atomic_fetch_add(p, v, __ATOMIC_RELAXED, __HIP_MEMORY_SCOPE_AGENT);
}

// --------------------- Tensor Data Mover tile load -------------------------
// Loads a (tileRows x 32) bf16 tile (row pitch strideK halfs in global) into
// LDS at ldsByteOff with pad_interval=16 DW / pad_amount=4 DW => LDS pitch of
// 40 halfs (80 B, 16B-aligned fragment runs). Rows >= validRows zero-fill
// (tensor_dim1 OOB), which pads the partial last M-block for free.
__device__ __forceinline__ void tdm_load_tile_bf16(
        const unsigned short* gptr, unsigned ldsByteOff,
        int validRows, int tileRows, int strideK)
{
    unsigned long long ga = (unsigned long long)(uintptr_t)gptr;
    u32x4 g0;
    g0[0] = 1u;                                            // count=1 (valid, user)
    g0[1] = ldsByteOff;                                    // lds_addr
    g0[2] = (unsigned)(ga & 0xFFFFFFFFull);                // global_addr[31:0]
    g0[3] = (unsigned)((ga >> 32) & 0x01FFFFFFull)         // global_addr[56:32]
          | 0x80000000u;                                   // type = 2 ("image")
    i32x8 g1;
    g1[0] = (int)(0x00010000u          // data_size = 1 (2 bytes)
                | (1u << 20)           // pad_enable
                | (3u << 22)           // pad_interval: 16 DWORDs (one 64B row)
                | (3u << 25));         // pad_amount:   4 DWORDs (8 halfs)
    g1[1] = (int)(32u << 16);                              // tensor_dim0 = 32
    g1[2] = (int)(((unsigned)validRows & 0xFFFFu) << 16);  // tensor_dim1 lo
    g1[3] = (int)((((unsigned)validRows >> 16) & 0xFFFFu)
                | (32u << 16));                            // td1 hi | tile_dim0=32
    g1[4] = (int)((unsigned)tileRows & 0xFFFFu);           // tile_dim1 | tile_dim2=0
    g1[5] = (int)(unsigned)strideK;                        // tensor_dim0_stride lo
    g1[6] = 0;
    g1[7] = 0;
    i32x4 z4 = {0, 0, 0, 0};
#if __clang_major__ >= 23
    i32x8 z8 = {0, 0, 0, 0, 0, 0, 0, 0};
    __builtin_amdgcn_tensor_load_to_lds(g0, g1, z4, z4, z8, 0);
#else
    __builtin_amdgcn_tensor_load_to_lds(g0, g1, z4, z4, 0);
#endif
}

// --------------------------- small utility kernels -------------------------

__global__ void gcn2_zero_u32(unsigned int* __restrict__ p, long long n) {
    long long i = (long long)blockIdx.x * blockDim.x + threadIdx.x;
    long long stride = (long long)gridDim.x * blockDim.x;
    for (; i < n; i += stride) p[i] = 0u;
}

__global__ void gcn2_cvt_bf16(const float* __restrict__ src,
                              unsigned short* __restrict__ dst, int n) {
    int i = blockIdx.x * blockDim.x + threadIdx.x;
    if (i < n) dst[i] = f32_to_bf16(src[i]);
}

// W[K][Ncol] f32 -> Wt[Ncol][K] bf16 (transposed for TDM/K-contiguous frags)
__global__ void gcn2_cvt_bf16_t(const float* __restrict__ W,
                                unsigned short* __restrict__ Wt,
                                int K, int Ncol) {
    int i = blockIdx.x * blockDim.x + threadIdx.x;
    if (i < K * Ncol) {
        int k = i / Ncol, n = i - k * Ncol;
        Wt[(size_t)n * K + k] = f32_to_bf16(W[i]);
    }
}

// Abf = bf16(0.9*agg + 0.1*h0), vectorized 4-wide. n must be multiple of 4.
__global__ void gcn2_combine_bf16(const float* __restrict__ agg,
                                  const float* __restrict__ h0,
                                  unsigned short* __restrict__ outb,
                                  long long n4) {
    long long i = (long long)blockIdx.x * blockDim.x + threadIdx.x;
    long long stride = (long long)gridDim.x * blockDim.x;
    for (; i < n4; i += stride) {
        float4 a = ((const float4*)agg)[i];
        float4 h = ((const float4*)h0)[i];
        ushort4 o;
        o.x = f32_to_bf16(0.9f * a.x + 0.1f * h.x);
        o.y = f32_to_bf16(0.9f * a.y + 0.1f * h.y);
        o.z = f32_to_bf16(0.9f * a.z + 0.1f * h.z);
        o.w = f32_to_bf16(0.9f * a.w + 0.1f * h.w);
        ((ushort4*)outb)[i] = o;
    }
}

__global__ void gcn2_deg(const int* __restrict__ col,
                         unsigned int* __restrict__ deg, int E) {
    int e = blockIdx.x * blockDim.x + threadIdx.x;
    if (e < E) atomicAdd(&deg[col[e]], 1u);
}

__global__ void gcn2_dinv(const unsigned int* __restrict__ deg,
                          float* __restrict__ dinv, int N) {
    int i = blockIdx.x * blockDim.x + threadIdx.x;
    if (i < N) dinv[i] = rsqrtf((float)deg[i] + 1.0f);  // +1 self-loop
}

// ------------------------- edge aggregation (H = 128) ----------------------
__global__ __launch_bounds__(256)
void gcn2_aggregate(const int* __restrict__ row, const int* __restrict__ col,
                    const float* __restrict__ dinv, const float* __restrict__ h,
                    float* __restrict__ agg, int E, int N) {
    const int lane = threadIdx.x & 31;
    long long item = ((long long)blockIdx.x * blockDim.x + threadIdx.x) >> 5;
    const long long stride = ((long long)gridDim.x * blockDim.x) >> 5;
    const long long total = (long long)E + N;
    for (; item < total; item += stride) {
        int r, c; float nrm;
        if (item < E) {
            r = row[item]; c = col[item];
            nrm = dinv[r] * dinv[c];
        } else {
            r = (int)(item - E); c = r;
            float d = dinv[r];
            nrm = d * d;
        }
        float4 v = *(const float4*)(h + (size_t)r * 128 + lane * 4);
        float* ap = agg + (size_t)c * 128 + lane * 4;
        atomAddF(ap + 0, nrm * v.x);
        atomAddF(ap + 1, nrm * v.y);
        atomAddF(ap + 2, nrm * v.z);
        atomAddF(ap + 3, nrm * v.w);
    }
}

// ------------- layer-1 GEMM: f32 A fused staging + TDM-fed W ---------------
// h0 = relu(x @ w1 + b1). x is read exactly once -> f32->bf16 fused into the
// A-tile staging (VALU), while wave 0 streams W tiles via the Tensor Data
// Mover into a ping-pong LDS buffer, overlapping DMA with staging/compute.
__global__ __launch_bounds__(256)
void gcn2_gemm_wmma_stage(const float* __restrict__ A,
                          const unsigned short* __restrict__ Wt, // bf16 [128][K]
                          const float* __restrict__ bias,
                          float* __restrict__ out, int M, int K) {
    constexpr int NT = 8, NCOL = 128, LDA = 40;
    __shared__ __align__(16) unsigned short sA[128 * LDA];
    __shared__ __align__(16) unsigned short sB[2][NCOL * LDA];

    const int tid = threadIdx.x, wave = tid >> 5, lane = tid & 31;
    const int blockRow = blockIdx.x * 128;

    v8f acc[NT];
    for (int t = 0; t < NT; ++t)
        for (int i = 0; i < 8; ++i) acc[t][i] = 0.0f;

    const int nK = K >> 5;
    if (wave == 0)
        tdm_load_tile_bf16(Wt, (unsigned)(uintptr_t)&sB[0][0], NCOL, NCOL, K);

    for (int kt = 0; kt < nK; ++kt) {
        const int buf = kt & 1;
        {   // stage A tile 128x32 f32 -> bf16 (all threads)
            const int r = tid >> 1, c0 = (tid & 1) << 4;
            const int gr = blockRow + r;
            alignas(16) unsigned short tmp[16];
            if (gr < M) {
                const float* src = A + (size_t)gr * K + kt * 32 + c0;
                #pragma unroll
                for (int i = 0; i < 16; ++i) tmp[i] = f32_to_bf16(src[i]);
            } else {
                #pragma unroll
                for (int i = 0; i < 16; ++i) tmp[i] = 0;
            }
            uint4* dst = (uint4*)&sA[r * LDA + c0];
            dst[0] = ((const uint4*)tmp)[0];
            dst[1] = ((const uint4*)tmp)[1];
        }
        if (wave == 0) {
            if (kt + 1 < nK) {
                tdm_load_tile_bf16(Wt + (kt + 1) * 32,
                                   (unsigned)(uintptr_t)&sB[buf ^ 1][0],
                                   NCOL, NCOL, K);
                __builtin_amdgcn_s_wait_tensorcnt(1);  // tile kt retired
            } else {
                __builtin_amdgcn_s_wait_tensorcnt(0);
            }
        }
        __syncthreads();
        {
            const int kbA = (lane >> 4) << 3;
            const int kbB = (lane >> 4) << 4;
            const int rA = (wave << 4) + (lane & 15);
            Frag16 af;
            af.q[0] = *(const uint4*)&sA[rA * LDA + kbA];
            af.q[1] = *(const uint4*)&sA[rA * LDA + 16 + kbA];
            #pragma unroll
            for (int nt = 0; nt < NT; ++nt) {
                const int rB = (nt << 4) + (lane & 15);
                Frag16 bf;
                bf.q[0] = *(const uint4*)&sB[buf][rB * LDA + kbA];
                bf.q[1] = *(const uint4*)&sB[buf][rB * LDA + 16 + kbA];
                acc[nt] = __builtin_amdgcn_wmma_f32_16x16x32_bf16(
                    false, af.v, false, bf.v, (short)0, acc[nt], false, false);
            }
        }
        __syncthreads();
    }

    const int nl = lane & 15, mhi = (lane >> 4) << 3;
    #pragma unroll
    for (int nt = 0; nt < NT; ++nt) {
        const int n = (nt << 4) + nl;
        const float b = bias[n];
        #pragma unroll
        for (int r = 0; r < 8; ++r) {
            const int gm = blockRow + (wave << 4) + mhi + r;
            if (gm < M)
                out[(size_t)gm * NCOL + n] = fmaxf(acc[nt][r] + b, 0.0f);
        }
    }
}

// -------------- TDM-fed double-buffered WMMA GEMM (bf16 A) -----------------
// out[M,NCOL] = Ab[M,K] @ Wt[NCOL,K]^T (+ bias). Wave 0 drives the Tensor
// Data Mover: per k-step it issues next A/W tiles into the ping-pong buffer,
// then s_wait_tensorcnt retires the current pair while DMA overlaps compute.
template <int NT, bool BIAS>
__global__ __launch_bounds__(256)
void gcn2_gemm_wmma_tdm(const unsigned short* __restrict__ Ab,  // bf16 [M][K]
                        const unsigned short* __restrict__ Wt,  // bf16 [NCOL][K]
                        const float* __restrict__ bias,
                        float* __restrict__ out, int M, int K) {
    constexpr int NCOL = NT * 16;
    constexpr int LDT = 40;   // halfs: TDM pad turns 32-half rows into 40-half pitch
    __shared__ __align__(16) unsigned short sA[2][128 * LDT];
    __shared__ __align__(16) unsigned short sB[2][NCOL * LDT];

    const int tid = threadIdx.x, wave = tid >> 5, lane = tid & 31;
    const int blockRow = blockIdx.x * 128;
    const int validRows = min(128, M - blockRow);

    v8f acc[NT];
    for (int t = 0; t < NT; ++t)
        for (int i = 0; i < 8; ++i) acc[t][i] = 0.0f;

    const int nK = K >> 5;
    if (wave == 0) {
        tdm_load_tile_bf16(Ab + (size_t)blockRow * K,
                           (unsigned)(uintptr_t)&sA[0][0], validRows, 128, K);
        tdm_load_tile_bf16(Wt, (unsigned)(uintptr_t)&sB[0][0], NCOL, NCOL, K);
    }
    for (int kt = 0; kt < nK; ++kt) {
        const int buf = kt & 1;
        if (wave == 0) {
            if (kt + 1 < nK) {
                tdm_load_tile_bf16(Ab + (size_t)blockRow * K + (kt + 1) * 32,
                                   (unsigned)(uintptr_t)&sA[buf ^ 1][0],
                                   validRows, 128, K);
                tdm_load_tile_bf16(Wt + (kt + 1) * 32,
                                   (unsigned)(uintptr_t)&sB[buf ^ 1][0],
                                   NCOL, NCOL, K);
                __builtin_amdgcn_s_wait_tensorcnt(2);  // current pair retired
            } else {
                __builtin_amdgcn_s_wait_tensorcnt(0);
            }
        }
        __syncthreads();
        {
            const int kbA = (lane >> 4) << 3;
            const int kbB = (lane >> 4) << 4;
            const int rA = (wave << 4) + (lane & 15);
            Frag16 af;
            af.q[0] = *(const uint4*)&sA[buf][rA * LDT + kbA];
            af.q[1] = *(const uint4*)&sA[buf][rA * LDT + 16 + kbA];
            #pragma unroll
            for (int nt = 0; nt < NT; ++nt) {
                const int rB = (nt << 4) + (lane & 15);
                Frag16 bf;
                bf.q[0] = *(const uint4*)&sB[buf][rB * LDT + kbA];
                bf.q[1] = *(const uint4*)&sB[buf][rB * LDT + 16 + kbA];
                acc[nt] = __builtin_amdgcn_wmma_f32_16x16x32_bf16(
                    false, af.v, false, bf.v, (short)0, acc[nt], false, false);
            }
        }
        __syncthreads();
    }

    const int nl = lane & 15, mhi = (lane >> 4) << 3;
    #pragma unroll
    for (int nt = 0; nt < NT; ++nt) {
        const int n = (nt << 4) + nl;
        const float b = BIAS ? bias[n] : 0.0f;
        #pragma unroll
        for (int r = 0; r < 8; ++r) {
            const int gm = blockRow + (wave << 4) + mhi + r;
            if (gm < M)
                out[(size_t)gm * NCOL + n] = acc[nt][r] + b;
        }
    }
}

// ------------------------------- host side ---------------------------------

extern "C" void kernel_launch(void* const* d_in, const int* in_sizes, int n_in,
                              void* d_out, int out_size, void* d_ws, size_t ws_size,
                              hipStream_t stream) {
    (void)n_in; (void)out_size; (void)ws_size;
    const int DIN = 512, H = 128, DOUT = 64, L = 2;

    const float* x  = (const float*)d_in[0];
    const int*   ei = (const int*)d_in[1];
    const float* w1 = (const float*)d_in[2];
    const float* b1 = (const float*)d_in[3];
    const float* cw = (const float*)d_in[4];
    const float* w2 = (const float*)d_in[5];
    const float* b2 = (const float*)d_in[6];

    const int N = in_sizes[0] / DIN;
    const int E = in_sizes[1] / 2;
    const int* rowIdx = ei;
    const int* colIdx = ei + E;

    float* out    = (float*)d_out;
    float* h_out  = out;                       // [N, H] (returned h; also h1 temp)
    float* logits = out + (size_t)N * H;       // [N, DOUT]

    char* ws = (char*)d_ws;
    auto carve = [&](size_t bytes) {
        char* p = ws;
        ws += (bytes + 255) & ~(size_t)255;
        return p;
    };
    float*          h0   = (float*)carve((size_t)N * H * 4);
    float*          agg  = (float*)carve((size_t)N * H * 4);
    unsigned short* Abf  = (unsigned short*)carve((size_t)N * H * 2);
    unsigned int*   deg  = (unsigned int*)carve((size_t)N * 4);
    float*          dinv = (float*)carve((size_t)N * 4);
    unsigned short* w1t  = (unsigned short*)carve((size_t)H * DIN * 2);     // [H][DIN]
    unsigned short* cwt  = (unsigned short*)carve((size_t)L * H * H * 2);   // [l][N][K]
    unsigned short* w2t  = (unsigned short*)carve((size_t)DOUT * H * 2);    // [N][K]

    const int T = 256;
    const dim3 gemmGrid((N + 127) / 128);

    // 1) weights -> bf16, transposed to [N][K]
    gcn2_cvt_bf16_t<<<(DIN * H + T - 1) / T, T, 0, stream>>>(w1, w1t, DIN, H);
    gcn2_cvt_bf16_t<<<(H * H + T - 1) / T, T, 0, stream>>>(cw, cwt, H, H);
    gcn2_cvt_bf16_t<<<(H * H + T - 1) / T, T, 0, stream>>>(cw + (size_t)H * H,
                                                           cwt + (size_t)H * H, H, H);
    gcn2_cvt_bf16_t<<<(H * DOUT + T - 1) / T, T, 0, stream>>>(w2, w2t, H, DOUT);

    // 2) degrees / dinv
    gcn2_zero_u32<<<1024, T, 0, stream>>>(deg, (long long)N);
    gcn2_deg<<<(E + T - 1) / T, T, 0, stream>>>(colIdx, deg, E);
    gcn2_dinv<<<(N + T - 1) / T, T, 0, stream>>>(deg, dinv, N);

    // 3) h0 = relu(x @ w1 + b1)
    gcn2_gemm_wmma_stage<<<gemmGrid, T, 0, stream>>>(x, w1t, b1, h0, N, DIN);

    // 4) layer 0: h1 = bf16(0.9*Ahat@h0 + 0.1*h0) @ cw[0]   (h1 parked in d_out)
    gcn2_zero_u32<<<8192, T, 0, stream>>>((unsigned int*)agg, (long long)N * H);
    gcn2_aggregate<<<4096, T, 0, stream>>>(rowIdx, colIdx, dinv, h0, agg, E, N);
    gcn2_combine_bf16<<<4096, T, 0, stream>>>(agg, h0, Abf, (long long)N * H / 4);
    gcn2_gemm_wmma_tdm<8, false><<<gemmGrid, T, 0, stream>>>(Abf, cwt, nullptr,
                                                             h_out, N, H);

    //    layer 1: h = bf16(0.9*Ahat@h1 + 0.1*h0) @ cw[1] -> d_out (over h1)
    gcn2_zero_u32<<<8192, T, 0, stream>>>((unsigned int*)agg, (long long)N * H);
    gcn2_aggregate<<<4096, T, 0, stream>>>(rowIdx, colIdx, dinv, h_out, agg, E, N);
    gcn2_combine_bf16<<<4096, T, 0, stream>>>(agg, h0, Abf, (long long)N * H / 4);
    gcn2_gemm_wmma_tdm<8, false><<<gemmGrid, T, 0, stream>>>(Abf,
                                                             cwt + (size_t)H * H,
                                                             nullptr, h_out, N, H);

    // 5) logits = bf16(h) @ w2 + b2
    gcn2_cvt_bf16<<<(N * H + T - 1) / T, T, 0, stream>>>(h_out, Abf, N * H);
    gcn2_gemm_wmma_tdm<4, true><<<gemmGrid, T, 0, stream>>>(Abf, w2t, b2,
                                                            logits, N, H);
}
